// LSTMDecoder_69535520522443
// MI455X (gfx1250) — matrix operand order repaired and checked by reading, not compiled, not run on hardware
//
#include <hip/hip_runtime.h>
#include <hip/hip_bf16.h>
#include <math.h>

// Problem constants (from reference)
#define BB  128
#define LL  128
#define HE  1024
#define HD  1024
#define EE  512
#define AP  512
#define TT  64
#define K1  (EE + HE + HD)   // 2560  [x | a | h1] for cell 1
#define K2  (HD + HD)        // 2048  [h1n | h2]   for cell 2
#define G4  (4 * HD)         // 4096 gate width

typedef __attribute__((ext_vector_type(16))) __bf16 v16bf;
typedef __attribute__((ext_vector_type(8)))  __bf16 bf16x8;
typedef __attribute__((ext_vector_type(8)))  float  v8f;

static __device__ __forceinline__ __bf16 f2bf(float f) {
    unsigned u = __builtin_bit_cast(unsigned, f);
    unsigned r = (u + 0x7FFFu + ((u >> 16) & 1u)) >> 16;
    unsigned short s = (unsigned short)r;
    return __builtin_bit_cast(__bf16, s);
}
static __device__ __forceinline__ float sigm(float x) { return 1.0f / (1.0f + expf(-x)); }

// ---------------------------------------------------------------------------
// Convert fp32 -> bf16 with optional row restride (dst_ld >= ncols)
__global__ void k_cvt_bf16(const float* __restrict__ src, __bf16* __restrict__ dst,
                           int dst_ld, int ncols, int total) {
    int idx = blockIdx.x * blockDim.x + threadIdx.x;
    if (idx >= total) return;
    int b = idx / ncols, j = idx - b * ncols;
    dst[(size_t)b * dst_ld + j] = f2bf(src[idx]);
}

// Transpose-convert: src (Kin x Nin fp32, row-major) -> dst[n*ldout + koff + k] bf16
__global__ void k_tr_bf16(const float* __restrict__ src, int Kin, int Nin,
                          __bf16* __restrict__ dst, int ldout, int koff) {
    int idx = blockIdx.x * blockDim.x + threadIdx.x;
    if (idx >= Kin * Nin) return;
    int k = idx / Nin, n = idx - k * Nin;
    dst[(size_t)n * ldout + koff + k] = f2bf(src[idx]);
}

__global__ void k_fill_bf16(__bf16* __restrict__ dst, float val, int n) {
    int idx = blockIdx.x * blockDim.x + threadIdx.x;
    if (idx < n) dst[idx] = f2bf(val);
}

__global__ void k_bias_add(const float* __restrict__ a, const float* __restrict__ b,
                           float* __restrict__ out, int n) {
    int idx = blockIdx.x * blockDim.x + threadIdx.x;
    if (idx < n) out[idx] = a[idx] + b[idx];
}

// ---------------------------------------------------------------------------
// bf16 WMMA GEMM:  C(MxN,f32,ldc) = act( A(MxK bf16 row-major, lda) * Bt^T + bias )
// Bt is N x K bf16 row-major (pre-transposed weight).
// 256 threads = 8 waves. Each wave computes a 64x16 tile: 4 M-tiles x 1 N-tile
// with 4 f32 accumulators, reusing one B fragment per K-step (4 WMMA : 5 loads).
// The 8 waves of a block cover 8 adjacent N-tiles over the SAME 64 rows, so the
// A fragments are identical across waves and hit the WGP$ after the first wave.
// Requires M % 64 == 0, N % 128 == 0, K % 32 == 0 (all shapes here comply).
__global__ void k_gemm_bf16(const __bf16* __restrict__ A, int lda,
                            const __bf16* __restrict__ Bt,
                            const float* __restrict__ bias,
                            float* __restrict__ C, int ldc,
                            int M, int N, int K, int act) {
    const int lane  = threadIdx.x & 31;
    const int wave  = threadIdx.x >> 5;
    const int ntile = blockIdx.x * 8 + wave;
    const int mbase = blockIdx.y * 64;
    if (ntile * 16 >= N || mbase >= M) return;   // wave-uniform: EXEC stays all-ones

    const int g  = lane >> 4;                    // half-wave group (0/1)
    const int ml = lane & 15;

    const __bf16* __restrict__ Arow0 = A + (size_t)(mbase + 0 * 16 + ml) * lda;
    const __bf16* __restrict__ Arow1 = A + (size_t)(mbase + 1 * 16 + ml) * lda;
    const __bf16* __restrict__ Arow2 = A + (size_t)(mbase + 2 * 16 + ml) * lda;
    const __bf16* __restrict__ Arow3 = A + (size_t)(mbase + 3 * 16 + ml) * lda;
    const __bf16* __restrict__ Brow  = Bt + (size_t)(ntile * 16 + ml) * K;

    v8f acc0 = {}, acc1 = {}, acc2 = {}, acc3 = {};
    for (int k0 = 0; k0 < K; k0 += 32) {
        union { v16bf v; bf16x8 h[2]; } ub, ua0, ua1, ua2, ua3;
        // B 32x16 fragment from N-major transposed weight: halves 0-7 = K k0+8g..+7, halves 8-15 = +16
        ub.h[0]  = *(const bf16x8*)(Brow + k0 + g * 8);
        ub.h[1]  = *(const bf16x8*)(Brow + k0 + 16 + g * 8);
        // A 16x32 fragments: halves 0-7 = K k0+8g..+7, halves 8-15 = +16
        ua0.h[0] = *(const bf16x8*)(Arow0 + k0 + g * 8);
        ua0.h[1] = *(const bf16x8*)(Arow0 + k0 + 16 + g * 8);
        ua1.h[0] = *(const bf16x8*)(Arow1 + k0 + g * 8);
        ua1.h[1] = *(const bf16x8*)(Arow1 + k0 + 16 + g * 8);
        ua2.h[0] = *(const bf16x8*)(Arow2 + k0 + g * 8);
        ua2.h[1] = *(const bf16x8*)(Arow2 + k0 + 16 + g * 8);
        ua3.h[0] = *(const bf16x8*)(Arow3 + k0 + g * 8);
        ua3.h[1] = *(const bf16x8*)(Arow3 + k0 + 16 + g * 8);
        acc0 = __builtin_amdgcn_wmma_f32_16x16x32_bf16(false, ua0.v, false, ub.v, (short)0, acc0, false, false);
        acc1 = __builtin_amdgcn_wmma_f32_16x16x32_bf16(false, ua1.v, false, ub.v, (short)0, acc1, false, false);
        acc2 = __builtin_amdgcn_wmma_f32_16x16x32_bf16(false, ua2.v, false, ub.v, (short)0, acc2, false, false);
        acc3 = __builtin_amdgcn_wmma_f32_16x16x32_bf16(false, ua3.v, false, ub.v, (short)0, acc3, false, false);
    }

    const int col = ntile * 16 + ml;
    const float bv = bias ? bias[col] : 0.0f;
    v8f accs[4] = {acc0, acc1, acc2, acc3};
    #pragma unroll
    for (int i = 0; i < 4; ++i) {
        const int rbase = mbase + i * 16 + g * 8;
        #pragma unroll
        for (int r = 0; r < 8; ++r) {
            float v = accs[i][r] + bv;
            if (act) v = tanhf(v);
            C[(size_t)(rbase + r) * ldc + col] = v;
        }
    }
}

// ---------------------------------------------------------------------------
// Attention + softmax + context, and builds [x_t | a_t] bf16 into cat1.
// One block (256 thr = 8 waves) per batch row.
__global__ void k_attention(const float* __restrict__ enc,        // (B,L,HE) fp32
                            const float* __restrict__ enc_proj,   // (B,L,AP) fp32
                            const float* __restrict__ q,          // (B,AP)
                            const float* __restrict__ corr_w,     // (AP)
                            const float* __restrict__ corr_b,     // scalar
                            const int*   __restrict__ mask,       // (B,L)
                            const float* __restrict__ x_t,        // (B,E)
                            __bf16* __restrict__ cat1)            // (B,K1) bf16
{
    __shared__ float e_s[LL];
    const int b    = blockIdx.x;
    const int tid  = threadIdx.x;
    const int lane = tid & 31;
    const int wave = tid >> 5;

    // stage x_t into cat1[:, 0:E]
    for (int j = tid; j < EE; j += 256)
        cat1[(size_t)b * K1 + j] = f2bf(x_t[(size_t)b * EE + j]);

    // e_l = sum_p tanh(enc_proj + q) * corr_w  (+ corr_b), masked
    const float cb = corr_b[0];
    for (int l = wave; l < LL; l += 8) {
        const float* ep = enc_proj + ((size_t)b * LL + l) * AP;
        float acc = 0.0f;
        for (int p = lane; p < AP; p += 32)
            acc += tanhf(ep[p] + q[(size_t)b * AP + p]) * corr_w[p];
        #pragma unroll
        for (int off = 16; off > 0; off >>= 1)
            acc += __shfl_xor(acc, off, 32);
        if (lane == 0)
            e_s[l] = mask[(size_t)b * LL + l] ? -__builtin_inff() : (acc + cb);
    }
    __syncthreads();

    // softmax over L (wave 0 only; wave-uniform branch)
    if (tid < 32) {
        float mx = -__builtin_inff();
        for (int l = lane; l < LL; l += 32) mx = fmaxf(mx, e_s[l]);
        #pragma unroll
        for (int off = 16; off > 0; off >>= 1) mx = fmaxf(mx, __shfl_xor(mx, off, 32));
        float s = 0.0f;
        for (int l = lane; l < LL; l += 32) s += expf(e_s[l] - mx);
        #pragma unroll
        for (int off = 16; off > 0; off >>= 1) s += __shfl_xor(s, off, 32);
        const float inv = 1.0f / s;
        for (int l = lane; l < LL; l += 32) e_s[l] = expf(e_s[l] - mx) * inv;
    }
    __syncthreads();

    // a_t[h] = sum_l alpha_l * enc[b,l,h] -> cat1[:, E:E+HE] bf16
    for (int h = tid; h < HE; h += 256) {
        float acc = 0.0f;
        const float* er = enc + (size_t)b * LL * HE + h;
        for (int l = 0; l < LL; ++l)
            acc += e_s[l] * er[(size_t)l * HE];
        cat1[(size_t)b * K1 + EE + h] = f2bf(acc);
    }
}

// ---------------------------------------------------------------------------
// LSTM cell pointwise: gates (B,4H) fp32, c (B,H) fp32 in/out; writes bf16 h
// to two strided destinations (next-step GEMM activation slots).
__global__ void k_lstm_cell(const float* __restrict__ gates,
                            float* __restrict__ c,
                            __bf16* __restrict__ h1dst, int ld1,
                            __bf16* __restrict__ h2dst, int ld2) {
    int idx = blockIdx.x * blockDim.x + threadIdx.x;
    if (idx >= BB * HD) return;
    int b = idx / HD, j = idx - b * HD;
    const float* gr = gates + (size_t)b * G4;
    float gi = gr[j], gf = gr[HD + j], gg = gr[2 * HD + j], go = gr[3 * HD + j];
    float cn = sigm(gf) * c[idx] + sigm(gi) * tanhf(gg);
    float hn = sigm(go) * tanhf(cn);
    c[idx] = cn;
    __bf16 hb = f2bf(hn);
    h1dst[(size_t)b * ld1 + j] = hb;
    h2dst[(size_t)b * ld2 + j] = hb;
}

// ---------------------------------------------------------------------------
static inline size_t alignup(size_t x) { return (x + 255) & ~(size_t)255; }

extern "C" void kernel_launch(void* const* d_in, const int* in_sizes, int n_in,
                              void* d_out, int out_size, void* d_ws, size_t ws_size,
                              hipStream_t stream) {
    (void)in_sizes; (void)n_in; (void)out_size; (void)ws_size;
    const float* enc      = (const float*)d_in[0];
    const int*   masks    = (const int*)  d_in[1];
    const float* h1_0     = (const float*)d_in[2];
    const float* c1_0     = (const float*)d_in[3];
    const float* h2_0     = (const float*)d_in[4];
    const float* c2_0     = (const float*)d_in[5];
    const float* captions = (const float*)d_in[6];
    const float* att_W    = (const float*)d_in[7];
    const float* att_b    = (const float*)d_in[8];
    const float* dhp_W    = (const float*)d_in[9];
    const float* dhp_b    = (const float*)d_in[10];
    const float* corr_w   = (const float*)d_in[11];
    const float* corr_b   = (const float*)d_in[12];
    const float* lt_W     = (const float*)d_in[13];
    const float* lt_b     = (const float*)d_in[14];
    const float* W_ih1    = (const float*)d_in[15];
    const float* W_hh1    = (const float*)d_in[16];
    const float* b_ih1    = (const float*)d_in[17];
    const float* b_hh1    = (const float*)d_in[18];
    const float* W_ih2    = (const float*)d_in[19];
    const float* W_hh2    = (const float*)d_in[20];
    const float* b_ih2    = (const float*)d_in[21];
    const float* b_hh2    = (const float*)d_in[22];
    float* out = (float*)d_out;

    // ---- workspace carve-out ----
    char* p = (char*)d_ws;
    auto carve = [&](size_t bytes) { char* r = p; p += alignup(bytes); return r; };
    __bf16* enc_bf   = (__bf16*)carve((size_t)BB * LL * HE * 2);
    __bf16* att_Wt   = (__bf16*)carve((size_t)AP * HE * 2);
    __bf16* dhp_Wt   = (__bf16*)carve((size_t)AP * HD * 2);
    __bf16* Wcat1t   = (__bf16*)carve((size_t)G4 * K1 * 2);
    __bf16* Wcat2t   = (__bf16*)carve((size_t)G4 * K2 * 2);
    __bf16* lt_Wt    = (__bf16*)carve((size_t)HD * HD * 2);
    float*  enc_proj = (float*) carve((size_t)BB * LL * AP * 4);
    float*  qbuf     = (float*) carve((size_t)BB * AP * 4);
    float*  gates    = (float*) carve((size_t)BB * G4 * 4);
    __bf16* cat1     = (__bf16*)carve((size_t)BB * K1 * 2);
    __bf16* cat2     = (__bf16*)carve((size_t)BB * K2 * 2);
    __bf16* hprev_bf = (__bf16*)carve((size_t)BB * HD * 2);
    float*  c1       = (float*) carve((size_t)BB * HD * 4);
    float*  c2       = (float*) carve((size_t)BB * HD * 4);
    float*  bias1    = (float*) carve((size_t)G4 * 4);
    float*  bias2    = (float*) carve((size_t)G4 * 4);

    const int CT = 256;
    auto blocks = [](long long n, int t) { return (unsigned)((n + t - 1) / t); };

    // ---- one-time weight/activation prep ----
    k_cvt_bf16<<<blocks((long long)BB*LL*HE, CT), CT, 0, stream>>>(enc, enc_bf, HE, HE, BB*LL*HE);
    k_tr_bf16 <<<blocks((long long)HE*AP, CT), CT, 0, stream>>>(att_W, HE, AP, att_Wt, HE, 0);
    k_tr_bf16 <<<blocks((long long)HD*AP, CT), CT, 0, stream>>>(dhp_W, HD, AP, dhp_Wt, HD, 0);
    k_tr_bf16 <<<blocks((long long)(EE+HE)*G4, CT), CT, 0, stream>>>(W_ih1, EE+HE, G4, Wcat1t, K1, 0);
    k_tr_bf16 <<<blocks((long long)HD*G4, CT), CT, 0, stream>>>(W_hh1, HD, G4, Wcat1t, K1, EE+HE);
    k_tr_bf16 <<<blocks((long long)HD*G4, CT), CT, 0, stream>>>(W_ih2, HD, G4, Wcat2t, K2, 0);
    k_tr_bf16 <<<blocks((long long)HD*G4, CT), CT, 0, stream>>>(W_hh2, HD, G4, Wcat2t, K2, HD);
    k_tr_bf16 <<<blocks((long long)HD*HD, CT), CT, 0, stream>>>(lt_W, HD, HD, lt_Wt, HD, 0);
    k_bias_add<<<blocks(G4, CT), CT, 0, stream>>>(b_ih1, b_hh1, bias1, G4);
    k_bias_add<<<blocks(G4, CT), CT, 0, stream>>>(b_ih2, b_hh2, bias2, G4);

    // state init: c fp32 copies; h states as bf16 in their GEMM slots
    hipMemcpyAsync(c1, c1_0, (size_t)BB*HD*4, hipMemcpyDeviceToDevice, stream);
    hipMemcpyAsync(c2, c2_0, (size_t)BB*HD*4, hipMemcpyDeviceToDevice, stream);
    k_cvt_bf16<<<blocks((long long)BB*HD, CT), CT, 0, stream>>>(h1_0, cat1 + (EE+HE), K1, HD, BB*HD);
    k_cvt_bf16<<<blocks((long long)BB*HD, CT), CT, 0, stream>>>(h2_0, cat2 + HD, K2, HD, BB*HD);
    k_fill_bf16<<<blocks((long long)BB*HD, CT), CT, 0, stream>>>(hprev_bf, 1.0f / (float)HE, BB*HD);

    // enc_proj = enc @ att_W + att_b   (M=B*L=16384, N=AP, K=HE)
    k_gemm_bf16<<<dim3(AP/128, (BB*LL)/64), CT, 0, stream>>>(
        enc_bf, HE, att_Wt, att_b, enc_proj, AP, BB*LL, AP, HE, 0);

    // ---- recurrence ----
    for (int t = 0; t < TT; ++t) {
        // q = h_prev @ dhp_W + dhp_b
        k_gemm_bf16<<<dim3(AP/128, BB/64), CT, 0, stream>>>(
            hprev_bf, HD, dhp_Wt, dhp_b, qbuf, AP, BB, AP, HD, 0);
        // attention -> cat1[:, 0:E]=x_t, cat1[:, E:E+HE]=a_t
        k_attention<<<BB, CT, 0, stream>>>(
            enc, enc_proj, qbuf, corr_w, corr_b, masks,
            captions + (size_t)t * BB * EE, cat1);
        // gates1 = [x|a|h1] @ [W_ih1;W_hh1] + b
        k_gemm_bf16<<<dim3(G4/128, BB/64), CT, 0, stream>>>(
            cat1, K1, Wcat1t, bias1, gates, G4, BB, G4, K1, 0);
        // cell1 -> h1n bf16 into cat2[:,0:HD] (for gates2/out) and cat1 h1 slot (next step)
        k_lstm_cell<<<blocks((long long)BB*HD, CT), CT, 0, stream>>>(
            gates, c1, cat2, K2, cat1 + (EE+HE), K1);
        // gates2 = [h1n|h2] @ [W_ih2;W_hh2] + b
        k_gemm_bf16<<<dim3(G4/128, BB/64), CT, 0, stream>>>(
            cat2, K2, Wcat2t, bias2, gates, G4, BB, G4, K2, 0);
        // cell2 -> h2n bf16 into cat2[:,HD:2HD] (next step) and hprev_bf
        k_lstm_cell<<<blocks((long long)BB*HD, CT), CT, 0, stream>>>(
            gates, c2, cat2 + HD, K2, hprev_bf, HD);
        // out_t = tanh(h1n @ lt_W + lt_b)  (A = cat2[:,0:HD], lda=K2)
        k_gemm_bf16<<<dim3(HD/128, BB/64), CT, 0, stream>>>(
            cat2, K2, lt_Wt, lt_b, out + (size_t)t * BB * HD, HD, BB, HD, HD, 1);
    }
}
